// DMPNN_678604832934
// MI455X (gfx1250) — matrix operand order repaired and checked by reading, not compiled, not run on hardware
//
#include <hip/hip_runtime.h>
#include <hip/hip_bf16.h>
#include <math.h>

// ---------------------------------------------------------------------------
// DMPNN forward for MI455X (gfx1250, wave32, WMMA).
// All heavy GEMMs run through v_wmma_f32_16x16x32_bf16 (bf16 in, fp32 acc).
// Tiling: one block = 4 waves = 16 rows x 256 cols; each wave owns a 64-col
// slice of the same 16 rows, so the A operand is read from HBM exactly once
// (wave-duplicate fragment loads hit WGP cache). Scatter/segment ops use
// native f32 global atomics.
// ---------------------------------------------------------------------------

typedef __attribute__((ext_vector_type(16))) __bf16 v16bf;
typedef __attribute__((ext_vector_type(8)))  float  v8f;

#define N_NODES 100000
#define N_EDGES 400000
#define N_GRAPH 4096
#define MOL_IN  25
#define F_AB    100
#define DIM     256

__device__ __forceinline__ float sigmoidf_(float x) { return 1.0f / (1.0f + __expf(-x)); }

// float atomic max via sign-split integer ordering (init value must be -inf)
__device__ __forceinline__ void atomicMaxF(float* addr, float v) {
    if (v >= 0.0f) atomicMax((int*)addr, __float_as_int(v));
    else           atomicMin((unsigned int*)addr, __float_as_uint(v));
}

// ---------------------------------------------------------------------------
// Generic WMMA GEMM:  C[M,NOUT] = act( A[M,K](lda) @ W[K,NOUT] + bias (+extra) )
// grid = (M/16, NOUT/256); block = 128 threads = 4 waves.
// Wave w computes rows [16*bx, 16*bx+16) x cols [(4*by+w)*64, +64).
// A fragment (16x32 bf16): lane l holds row (l&15); elem j -> K = 16*(j>>3) + 8*(l>>4) + (j&7)
// B fragment (32x16 bf16): lane l holds col N=(l&15); elem j -> K = 16*(j>>3) + 8*(l>>4) + (j&7)
// C/D (16x16 f32, v8f):    elem i -> row = i + 8*(l>>4), col = (l&15)
// ---------------------------------------------------------------------------
template<bool RELU, bool HAS_EXTRA>
__global__ void wmma_gemm(const float* __restrict__ A, int lda,
                          const float* __restrict__ W,
                          const float* __restrict__ bias,
                          const float* __restrict__ extra,
                          float* __restrict__ C,
                          int M, int K, int NOUT)
{
    const int lane = threadIdx.x & 31;
    const int wave = threadIdx.x >> 5;
    const int row0 = blockIdx.x * 16;
    if (row0 >= M) return;                       // wave-uniform (M % 16 == 0)
    const int n0 = (blockIdx.y * 4 + wave) * 64;

    v8f acc0 = {}, acc1 = {}, acc2 = {}, acc3 = {};

    const int   arow = row0 + (lane & 15);
    const int   kb   = (lane >> 4) << 3;         // 0 or 8
    const float* Arow = A + (long long)arow * lda;

    for (int k0 = 0; k0 < K; k0 += 32) {
        if (k0 + 32 < K) __builtin_prefetch(Arow + k0 + 32, 0, 1);  // global_prefetch_b8

        v16bf a;
        #pragma unroll
        for (int j = 0; j < 8; ++j) {
            int k1 = k0 + kb + j;
            int k2 = k0 + 16 + kb + j;
            a[j]     = (k1 < K) ? (__bf16)Arow[k1] : (__bf16)0.0f;
            a[8 + j] = (k2 < K) ? (__bf16)Arow[k2] : (__bf16)0.0f;
        }

        v16bf b0, b1, b2, b3;
        #pragma unroll
        for (int j = 0; j < 16; ++j) {
            const int kk = k0 + 16 * (j >> 3) + kb + (j & 7);
            const bool kv = kk < K;
            const float* Wrow = W + (long long)kk * NOUT + n0 + (lane & 15);
            b0[j] = kv ? (__bf16)Wrow[0]  : (__bf16)0.0f;
            b1[j] = kv ? (__bf16)Wrow[16] : (__bf16)0.0f;
            b2[j] = kv ? (__bf16)Wrow[32] : (__bf16)0.0f;
            b3[j] = kv ? (__bf16)Wrow[48] : (__bf16)0.0f;
        }

        acc0 = __builtin_amdgcn_wmma_f32_16x16x32_bf16(false, a, false, b0, (short)0, acc0, false, false);
        acc1 = __builtin_amdgcn_wmma_f32_16x16x32_bf16(false, a, false, b1, (short)0, acc1, false, false);
        acc2 = __builtin_amdgcn_wmma_f32_16x16x32_bf16(false, a, false, b2, (short)0, acc2, false, false);
        acc3 = __builtin_amdgcn_wmma_f32_16x16x32_bf16(false, a, false, b3, (short)0, acc3, false, false);
    }

    const int cr = (lane >> 4) << 3;
    const int cc = lane & 15;
    #pragma unroll
    for (int i = 0; i < 8; ++i) {
        const int row = row0 + cr + i;
        float* crow = C + (long long)row * NOUT + n0;
        const float* erow = HAS_EXTRA ? (extra + (long long)row * NOUT + n0) : nullptr;
        float v;
        v = acc0[i] + bias[n0 + cc];       if (HAS_EXTRA) v += erow[cc];       if (RELU) v = fmaxf(v, 0.0f); crow[cc]      = v;
        v = acc1[i] + bias[n0 + 16 + cc];  if (HAS_EXTRA) v += erow[16 + cc];  if (RELU) v = fmaxf(v, 0.0f); crow[16 + cc] = v;
        v = acc2[i] + bias[n0 + 32 + cc];  if (HAS_EXTRA) v += erow[32 + cc];  if (RELU) v = fmaxf(v, 0.0f); crow[32 + cc] = v;
        v = acc3[i] + bias[n0 + 48 + cc];  if (HAS_EXTRA) v += erow[48 + cc];  if (RELU) v = fmaxf(v, 0.0f); crow[48 + cc] = v;
    }
}

// ---------------------------------------------------------------------------
// Fused edge GEMM + message scatter:
//   ew = edge_attr @ nn_w + nn_b           (computed in registers, never stored)
//   agg[dst] += h0[src] * ew               (f32 atomics into agg)
// grid = (E/16); block = 4 waves; wave w -> cols [64w, 64w+64) of same 16 edges.
// edge_attr streamed from HBM exactly once.
// ---------------------------------------------------------------------------
__global__ void edge_gemm_scatter(const float* __restrict__ EA,
                                  const float* __restrict__ W,
                                  const float* __restrict__ bias,
                                  const int*   __restrict__ eidx,   // [2][E]
                                  const float* __restrict__ h0,
                                  float* __restrict__ agg)
{
    const int lane = threadIdx.x & 31;
    const int wave = threadIdx.x >> 5;
    const int row0 = blockIdx.x * 16;   // edge tile base
    const int n0   = wave * 64;
    const int K    = F_AB;

    v8f acc0 = {}, acc1 = {}, acc2 = {}, acc3 = {};

    const int   arow = row0 + (lane & 15);
    const int   kb   = (lane >> 4) << 3;
    const float* Arow = EA + (long long)arow * K;

    for (int k0 = 0; k0 < K; k0 += 32) {
        if (k0 + 32 < K) __builtin_prefetch(Arow + k0 + 32, 0, 1);
        v16bf a;
        #pragma unroll
        for (int j = 0; j < 8; ++j) {
            int k1 = k0 + kb + j;
            int k2 = k0 + 16 + kb + j;
            a[j]     = (k1 < K) ? (__bf16)Arow[k1] : (__bf16)0.0f;
            a[8 + j] = (k2 < K) ? (__bf16)Arow[k2] : (__bf16)0.0f;
        }
        v16bf b0, b1, b2, b3;
        #pragma unroll
        for (int j = 0; j < 16; ++j) {
            const int kk = k0 + 16 * (j >> 3) + kb + (j & 7);
            const bool kv = kk < K;
            const float* Wrow = W + (long long)kk * DIM + n0 + (lane & 15);
            b0[j] = kv ? (__bf16)Wrow[0]  : (__bf16)0.0f;
            b1[j] = kv ? (__bf16)Wrow[16] : (__bf16)0.0f;
            b2[j] = kv ? (__bf16)Wrow[32] : (__bf16)0.0f;
            b3[j] = kv ? (__bf16)Wrow[48] : (__bf16)0.0f;
        }
        acc0 = __builtin_amdgcn_wmma_f32_16x16x32_bf16(false, a, false, b0, (short)0, acc0, false, false);
        acc1 = __builtin_amdgcn_wmma_f32_16x16x32_bf16(false, a, false, b1, (short)0, acc1, false, false);
        acc2 = __builtin_amdgcn_wmma_f32_16x16x32_bf16(false, a, false, b2, (short)0, acc2, false, false);
        acc3 = __builtin_amdgcn_wmma_f32_16x16x32_bf16(false, a, false, b3, (short)0, acc3, false, false);
    }

    const int cr = (lane >> 4) << 3;
    const int cc = lane & 15;
    #pragma unroll
    for (int i = 0; i < 8; ++i) {
        const int e = row0 + cr + i;
        const int s = eidx[e];              // src
        const int d = eidx[N_EDGES + e];    // dst
        const float* hs = h0  + (long long)s * DIM + n0;
        float*       ad = agg + (long long)d * DIM + n0;
        float v;
        v = acc0[i] + bias[n0 + cc];      atomicAdd(&ad[cc],      hs[cc]      * v);
        v = acc1[i] + bias[n0 + 16 + cc]; atomicAdd(&ad[16 + cc], hs[16 + cc] * v);
        v = acc2[i] + bias[n0 + 32 + cc]; atomicAdd(&ad[32 + cc], hs[32 + cc] * v);
        v = acc3[i] + bias[n0 + 48 + cc]; atomicAdd(&ad[48 + cc], hs[48 + cc] * v);
    }
}

// ---------------------------------------------------------------------------
// Small helper kernels
// ---------------------------------------------------------------------------
__global__ void zero_kernel(float* p, long long n) {
    long long i = (long long)blockIdx.x * 256 + threadIdx.x;
    if (i < n) p[i] = 0.0f;
}

__global__ void build_wcat(const float* __restrict__ wih, const float* __restrict__ whh,
                           float* __restrict__ wcat) {
    int i = blockIdx.x * 256 + threadIdx.x;        // 768*1024 elements
    if (i >= 768 * 1024) return;
    int row = i >> 10, col = i & 1023;
    wcat[i] = (row < 512) ? wih[i] : whh[(row - 512) * 1024 + col];
}

// LSTM cell update + prep next attention round (reset r accumulator, m, z)
__global__ void lstm_update(const float* __restrict__ gates,
                            float* __restrict__ cst, float* __restrict__ qcat,
                            float* __restrict__ m, float* __restrict__ z) {
    int i = blockIdx.x * 256 + threadIdx.x;        // B*DIM elements
    if (i >= N_GRAPH * DIM) return;
    int b = i >> 8, d = i & 255;
    const float* g = gates + (long long)b * 1024;
    float gi = g[d], gf = g[DIM + d], gg = g[2 * DIM + d], go = g[3 * DIM + d];
    float c = sigmoidf_(gf) * cst[i] + sigmoidf_(gi) * tanhf(gg);
    cst[i] = c;
    float hq = sigmoidf_(go) * tanhf(c);
    float* q = qcat + (long long)b * 768;
    q[d]       = hq;    // q_star hq half
    q[512 + d] = hq;    // hq slot (for whh)
    q[256 + d] = 0.0f;  // reset r accumulator
    if (d == 0) { m[b] = -__builtin_inff(); z[b] = 0.0f; }
}

// e[n] = <h[n], hq[batch[n]]> ; segment max via atomics. One wave per node.
__global__ void escore(const float* __restrict__ h, const float* __restrict__ qcat,
                       const int* __restrict__ batch,
                       float* __restrict__ ebuf, float* __restrict__ m) {
    int lane = threadIdx.x & 31, wave = threadIdx.x >> 5;
    int n = blockIdx.x * 8 + wave;
    if (n >= N_NODES) return;
    int b = batch[n];
    const float* hr = h + (long long)n * DIM;
    const float* hq = qcat + (long long)b * 768 + 512;
    float s = 0.0f;
    #pragma unroll
    for (int j = 0; j < 8; ++j) { int d = lane + 32 * j; s += hr[d] * hq[d]; }
    #pragma unroll
    for (int off = 16; off > 0; off >>= 1) s += __shfl_xor(s, off, 32);
    if (lane == 0) { ebuf[n] = s; atomicMaxF(&m[b], s); }
}

__global__ void expsum(const float* __restrict__ ebuf, const int* __restrict__ batch,
                       const float* __restrict__ m,
                       float* __restrict__ abuf, float* __restrict__ z) {
    int n = blockIdx.x * 256 + threadIdx.x;
    if (n >= N_NODES) return;
    int b = batch[n];
    float a = __expf(ebuf[n] - m[b]);
    abuf[n] = a;
    atomicAdd(&z[b], a);
}

// r[b] += (a[n]/z[b]) * h[n]   (writes into qcat[:,256:512])
__global__ void scatter_r(const float* __restrict__ h, const float* __restrict__ abuf,
                          const float* __restrict__ z, const int* __restrict__ batch,
                          float* __restrict__ qcat) {
    int n = blockIdx.x, d = threadIdx.x;
    int b = batch[n];
    float w = abuf[n] / z[b];
    atomicAdd(&qcat[(long long)b * 768 + 256 + d], w * h[(long long)n * DIM + d]);
}

// out[b] = <t[b], lin2_w> + lin2_b    (one wave per graph)
__global__ void head_out(const float* __restrict__ t, const float* __restrict__ w2,
                         const float* __restrict__ b2, float* __restrict__ out) {
    int lane = threadIdx.x & 31, wave = threadIdx.x >> 5;
    int b = blockIdx.x * 8 + wave;
    if (b >= N_GRAPH) return;
    const float* tr = t + (long long)b * DIM;
    float s = 0.0f;
    #pragma unroll
    for (int j = 0; j < 8; ++j) { int d = lane + 32 * j; s += tr[d] * w2[d]; }
    #pragma unroll
    for (int off = 16; off > 0; off >>= 1) s += __shfl_xor(s, off, 32);
    if (lane == 0) out[b] = s + b2[0];
}

// ---------------------------------------------------------------------------
extern "C" void kernel_launch(void* const* d_in, const int* in_sizes, int n_in,
                              void* d_out, int out_size, void* d_ws, size_t ws_size,
                              hipStream_t stream) {
    const float* x      = (const float*)d_in[0];
    const int*   eidx   = (const int*)  d_in[1];
    const float* eattr  = (const float*)d_in[2];
    const int*   batch  = (const int*)  d_in[3];
    const float* lin0_w = (const float*)d_in[4];
    const float* lin0_b = (const float*)d_in[5];
    const float* root_w = (const float*)d_in[6];
    const float* root_b = (const float*)d_in[7];
    const float* nn_w   = (const float*)d_in[8];
    const float* nn_b   = (const float*)d_in[9];
    const float* wih    = (const float*)d_in[10];
    const float* whh    = (const float*)d_in[11];
    const float* lstm_b = (const float*)d_in[12];
    const float* lin1_w = (const float*)d_in[13];
    const float* lin1_b = (const float*)d_in[14];
    const float* lin2_w = (const float*)d_in[15];
    const float* lin2_b = (const float*)d_in[16];
    float* out = (float*)d_out;

    // workspace layout (fp32)
    float* p    = (float*)d_ws;
    float* h0   = p; p += (long long)N_NODES * DIM;   // relu(x@lin0)
    float* agg  = p; p += (long long)N_NODES * DIM;   // message agg; becomes h in-place
    float* gates= p; p += (long long)N_GRAPH * 1024;
    float* qcat = p; p += (long long)N_GRAPH * 768;   // [hq | r | hq]
    float* cst  = p; p += (long long)N_GRAPH * DIM;
    float* wcat = p; p += 768LL * 1024;
    float* ebuf = p; p += N_NODES;
    float* abuf = p; p += N_NODES;
    float* mbuf = p; p += N_GRAPH;
    float* zbuf = p; p += N_GRAPH;
    float* tbuf = p; p += (long long)N_GRAPH * DIM;

    dim3 blk(128);

    // h0 = relu(x @ lin0_w + lin0_b)
    wmma_gemm<true, false><<<dim3(N_NODES / 16, 1), blk, 0, stream>>>(
        x, MOL_IN, lin0_w, lin0_b, nullptr, h0, N_NODES, MOL_IN, DIM);

    // agg = 0 ; agg[dst] += h0[src] * (edge_attr @ nn_w + nn_b)
    zero_kernel<<<(unsigned)(((long long)N_NODES * DIM + 255) / 256), 256, 0, stream>>>(
        agg, (long long)N_NODES * DIM);
    edge_gemm_scatter<<<dim3(N_EDGES / 16), blk, 0, stream>>>(
        eattr, nn_w, nn_b, eidx, h0, agg);

    // h = relu(h0 @ root_w + root_b + agg)   (in-place over agg: block owns its rows)
    wmma_gemm<true, true><<<dim3(N_NODES / 16, 1), blk, 0, stream>>>(
        h0, DIM, root_w, root_b, agg, agg, N_NODES, DIM, DIM);
    float* h = agg;

    // Set2Set state
    build_wcat<<<(768 * 1024 + 255) / 256, 256, 0, stream>>>(wih, whh, wcat);
    zero_kernel<<<(unsigned)(((long long)N_GRAPH * (768 + DIM) + 255) / 256), 256, 0, stream>>>(
        qcat, (long long)N_GRAPH * (768 + DIM));   // qcat + cst are adjacent

    for (int step = 0; step < 3; ++step) {
        wmma_gemm<false, false><<<dim3(N_GRAPH / 16, 1024 / 256), blk, 0, stream>>>(
            qcat, 768, wcat, lstm_b, nullptr, gates, N_GRAPH, 768, 1024);
        lstm_update<<<(N_GRAPH * DIM + 255) / 256, 256, 0, stream>>>(gates, cst, qcat, mbuf, zbuf);
        escore<<<(N_NODES + 7) / 8, 256, 0, stream>>>(h, qcat, batch, ebuf, mbuf);
        expsum<<<(N_NODES + 255) / 256, 256, 0, stream>>>(ebuf, batch, mbuf, abuf, zbuf);
        scatter_r<<<N_NODES, DIM, 0, stream>>>(h, abuf, zbuf, batch, qcat);
    }

    // t = relu(q_star @ lin1_w + lin1_b) ; out = t @ lin2_w + lin2_b
    wmma_gemm<true, false><<<dim3(N_GRAPH / 16, 1), blk, 0, stream>>>(
        qcat, 768, lin1_w, lin1_b, nullptr, tbuf, N_GRAPH, 512, DIM);
    head_out<<<(N_GRAPH + 7) / 8, 256, 0, stream>>>(tbuf, lin2_w, lin2_b, out);
}
